// PytorchLlamaAttentionModel_13434657702591
// MI455X (gfx1250) — matrix operand order repaired and checked by reading, not compiled, not run on hardware
//
#include <hip/hip_runtime.h>
#include <math.h>

// LLama attention for MI455X (gfx1250, wave32, WMMA).
// All GEMM-like math in bf16 WMMA (v_wmma_f32_16x16x32_bf16), f32 accumulate.
// fp32 sources converted to bf16 while staging to LDS (conversion VALU
// co-executes with WMMA since 16-bit WMMA is TRANS-tracked).
// K-tiles in attention staged with GLOBAL_LOAD_ASYNC_TO_LDS_B128 (ASYNCcnt).
// Inner loops use explicit double-buffered fragment pipelining so each
// s_wait_dscnt covers a ds_load issued one WMMA earlier.
// Workspace (needs 64 MB): q | k | v | attn, each S*D bf16.

typedef unsigned short u16;
typedef unsigned int   u32;

typedef __attribute__((ext_vector_type(16))) __bf16 v16bf;
typedef __attribute__((ext_vector_type(8)))  float  v8f;
typedef int v4i __attribute__((vector_size(16)));   // matches async-LDS builtin

union FragU { v16bf v; uint4 q[2]; };

static constexpr int S_  = 2048;
static constexpr int D_  = 4096;
static constexpr int NH_ = 32;
static constexpr int HD_ = 128;

#define AS1 __attribute__((address_space(1)))
#define AS3 __attribute__((address_space(3)))

#if defined(__has_builtin)
#  if __has_builtin(__builtin_amdgcn_global_load_async_to_lds_b128)
#    define HAVE_ASYNC_B128 1
#  else
#    define HAVE_ASYNC_B128 0
#  endif
#  if __has_builtin(__builtin_amdgcn_s_wait_asynccnt)
#    define HAVE_WAIT_ASYNC 1
#  else
#    define HAVE_WAIT_ASYNC 0
#  endif
#else
#  define HAVE_ASYNC_B128 0
#  define HAVE_WAIT_ASYNC 0
#endif

// async global -> LDS copy of 16 bytes (falls back to VGPR round-trip)
__device__ __forceinline__ void cp_async_b128(u16* lds_dst, const u16* gsrc) {
#if HAVE_ASYNC_B128
  __builtin_amdgcn_global_load_async_to_lds_b128(
      (AS1 v4i*)gsrc, (AS3 v4i*)lds_dst, 0, 0);
#else
  *(uint4*)lds_dst = *(const uint4*)gsrc;
#endif
}
__device__ __forceinline__ void wait_async0() {
#if HAVE_ASYNC_B128
#  if HAVE_WAIT_ASYNC
  __builtin_amdgcn_s_wait_asynccnt(0);
#  else
  asm volatile("s_wait_asynccnt 0x0" ::: "memory");
#  endif
#endif
}

__device__ __forceinline__ float bf2f(u16 h) {
  u32 u = ((u32)h) << 16; float f; __builtin_memcpy(&f, &u, 4); return f;
}
__device__ __forceinline__ u16 f2bf(float f) {
  u32 u; __builtin_memcpy(&u, &f, 4);
  u = (u + 0x7FFFu + ((u >> 16) & 1u)) >> 16;   // round-to-nearest-even
  return (u16)u;
}
__device__ __forceinline__ u32 pk2(float a, float b) {
  return (u32)f2bf(a) | ((u32)f2bf(b) << 16);
}
__device__ __forceinline__ v8f vzero() {
  v8f z = {0.f,0.f,0.f,0.f,0.f,0.f,0.f,0.f}; return z;
}
__device__ __forceinline__ float redmax16(float x) {
  x = fmaxf(x, __shfl_xor(x, 1, 32));
  x = fmaxf(x, __shfl_xor(x, 2, 32));
  x = fmaxf(x, __shfl_xor(x, 4, 32));
  x = fmaxf(x, __shfl_xor(x, 8, 32));
  return x;
}
__device__ __forceinline__ float redsum16(float x) {
  x += __shfl_xor(x, 1, 32);
  x += __shfl_xor(x, 2, 32);
  x += __shfl_xor(x, 4, 32);
  x += __shfl_xor(x, 8, 32);
  return x;
}

// ---------------------------------------------------------------------------
// GEMM: C[M,N] = A[M,K] * W[N,K]^T  (W fp32 row-major [N][K]).
// 128x128 block tile, 8 waves x (2 M-tiles x 4 N-tiles), TK=32.
// Double-buffered LDS + register-staged global loads; B-fragments pipelined.
// ---------------------------------------------------------------------------
static constexpr int TM = 128, TN = 128, TK = 32;
static constexpr int LDA_ = 40;   // halfs, padded (conflict-free ds_load_b128)
static constexpr int LDB_ = 40;

template<bool A_BF16, bool OUT_BF16>
__global__ __launch_bounds__(256)
void gemm_bt(const void* __restrict__ Ap, const float* __restrict__ W,
             void* __restrict__ Cp, int M, int N, int K)
{
  __shared__ u16 lA[2][TM * LDA_];
  __shared__ u16 lB[2][TN * LDB_];

  const int tid   = threadIdx.x;
  const int lane  = tid & 31;
  const int wave  = tid >> 5;
  const int wm    = wave >> 1;       // 0..3
  const int wn    = wave & 1;        // 0..1
  const int lhalf = lane & 15;
  const bool hi   = lane >= 16;
  const int bm    = blockIdx.x * TM;
  const int bn    = blockIdx.y * TN;

  const int sr  = tid & 127;         // staged row (A) / n-index (B)
  const int sc0 = (tid >> 7) * 16;   // k chunk: 0 or 16

  const u16*   A16 = (const u16*)Ap;
  const float* A32 = (const float*)Ap;

  uint4  aBf[2];     // A_BF16 staging regs
  float4 aFp[4];     // fp32-A staging regs
  float4 bFp[4];     // W staging regs

  v8f acc[2][4];
  for (int m = 0; m < 2; ++m)
    for (int n = 0; n < 4; ++n) acc[m][n] = vzero();

  auto gload = [&](int k0) {
    if (A_BF16) {
      const uint4* s = (const uint4*)(A16 + (size_t)(bm + sr) * K + k0 + sc0);
      aBf[0] = s[0]; aBf[1] = s[1];
    } else {
      const float4* s = (const float4*)(A32 + (size_t)(bm + sr) * K + k0 + sc0);
      aFp[0] = s[0]; aFp[1] = s[1]; aFp[2] = s[2]; aFp[3] = s[3];
    }
    const float4* t = (const float4*)(W + (size_t)(bn + sr) * K + k0 + sc0);
    bFp[0] = t[0]; bFp[1] = t[1]; bFp[2] = t[2]; bFp[3] = t[3];
  };
  auto lstore = [&](int buf) {
    if (A_BF16) {
      *(uint4*)&lA[buf][sr * LDA_ + sc0]     = aBf[0];
      *(uint4*)&lA[buf][sr * LDA_ + sc0 + 8] = aBf[1];
    } else {
      uint4 p0 = { pk2(aFp[0].x,aFp[0].y), pk2(aFp[0].z,aFp[0].w),
                   pk2(aFp[1].x,aFp[1].y), pk2(aFp[1].z,aFp[1].w) };
      uint4 p1 = { pk2(aFp[2].x,aFp[2].y), pk2(aFp[2].z,aFp[2].w),
                   pk2(aFp[3].x,aFp[3].y), pk2(aFp[3].z,aFp[3].w) };
      *(uint4*)&lA[buf][sr * LDA_ + sc0]     = p0;
      *(uint4*)&lA[buf][sr * LDA_ + sc0 + 8] = p1;
    }
    uint4 q0 = { pk2(bFp[0].x,bFp[0].y), pk2(bFp[0].z,bFp[0].w),
                 pk2(bFp[1].x,bFp[1].y), pk2(bFp[1].z,bFp[1].w) };
    uint4 q1 = { pk2(bFp[2].x,bFp[2].y), pk2(bFp[2].z,bFp[2].w),
                 pk2(bFp[3].x,bFp[3].y), pk2(bFp[3].z,bFp[3].w) };
    *(uint4*)&lB[buf][sr * LDB_ + sc0]     = q0;
    *(uint4*)&lB[buf][sr * LDB_ + sc0 + 8] = q1;
  };

  const int nk = K / TK;
  gload(0);
  int cur = 0;
  for (int kt = 0; kt < nk; ++kt) {
    lstore(cur);                            // waits in-flight global loads
    if (kt + 1 < nk) gload((kt + 1) * TK);  // overlaps with compute below
    __syncthreads();

    // A frag (16-bit 16x32): lanes 0-15: K 0-15; lanes 16-31: K 16-31
    FragU af[2];
    const int ka = hi ? 16 : 0;
    #pragma unroll
    for (int m = 0; m < 2; ++m) {
      const u16* p = &lA[cur][(wm*32 + m*16 + lhalf) * LDA_];
      af[m].q[0] = *(const uint4*)(p + ka);
      af[m].q[1] = *(const uint4*)(p + ka + 8);
    }
    // B frag (32x16): lanes 0-15 hold K 0-15, lanes 16-31 hold K 16-31.
    // Double-buffered: load frag n+1 before the WMMAs on frag n.
    const int kbb = hi ? 16 : 0;
    FragU fb[2];
    {
      const uint4* p = (const uint4*)&lB[cur][(wn*64 + lhalf) * LDB_ + kbb];
      fb[0].q[0] = p[0]; fb[0].q[1] = p[1];
    }
    #pragma unroll
    for (int n = 0; n < 4; ++n) {
      if (n < 3) {
        const uint4* p = (const uint4*)&lB[cur][(wn*64 + (n+1)*16 + lhalf) * LDB_ + kbb];
        fb[(n + 1) & 1].q[0] = p[0];
        fb[(n + 1) & 1].q[1] = p[1];
      }
      acc[0][n] = __builtin_amdgcn_wmma_f32_16x16x32_bf16(
          false, af[0].v, false, fb[n & 1].v, (short)0, acc[0][n], false, false);
      acc[1][n] = __builtin_amdgcn_wmma_f32_16x16x32_bf16(
          false, af[1].v, false, fb[n & 1].v, (short)0, acc[1][n], false, false);
    }
    cur ^= 1;
  }

  // C layout: VGPR v -> row = base + v + (hi?8:0), col = base + lane%16
  for (int m = 0; m < 2; ++m) {
    int rb = bm + wm*32 + m*16 + (hi ? 8 : 0);
    for (int n = 0; n < 4; ++n) {
      int col = bn + wn*64 + n*16 + lhalf;
      for (int vv = 0; vv < 8; ++vv) {
        float val = acc[m][n][vv];
        size_t idx = (size_t)(rb + vv) * N + col;
        if (OUT_BF16) ((u16*)Cp)[idx] = f2bf(val);
        else          ((float*)Cp)[idx] = val;
      }
    }
  }
}

// ---------------------------------------------------------------------------
// RoPE on bf16 q,k in place. One thread per (s, h, j<64): pair (j, j+64).
// ---------------------------------------------------------------------------
__global__ __launch_bounds__(256)
void rope_kernel(u16* __restrict__ q, u16* __restrict__ k,
                 const int* __restrict__ pos)
{
  int idx = blockIdx.x * blockDim.x + threadIdx.x;   // S*NH*64
  if (idx >= S_ * NH_ * 64) return;
  int s = idx >> 11;
  int r = idx & 2047;
  int h = r >> 6, j = r & 63;
  float p = (float)pos[s];
  float ang = p * exp2f(-(float)j * (13.287712379549449f / 64.0f)); // 10000^(-2j/128)
  float c, sn;
  __sincosf(ang, &c, &sn);
  size_t base = (size_t)s * D_ + h * HD_ + j;
  float a = bf2f(q[base]), b = bf2f(q[base + 64]);
  q[base]      = f2bf(a * c - b * sn);
  q[base + 64] = f2bf(b * c + a * sn);
  a = bf2f(k[base]); b = bf2f(k[base + 64]);
  k[base]      = f2bf(a * c - b * sn);
  k[base + 64] = f2bf(b * c + a * sn);
}

// ---------------------------------------------------------------------------
// Flash attention: block = (head, 128 query rows), 8 waves x 16 rows.
// Bc = 64 keys/iter; causal; online softmax; bf16 WMMA.
// ---------------------------------------------------------------------------
static constexpr int BR = 128, BC = 64;
static constexpr int LDK_ = HD_ + 8;  // lK[key][hd]
static constexpr int LDV_ = BC + 8;   // lV[hd][key]  (transposed V)
static constexpr int LDP_ = BC + 8;   // per-wave P scratch [16][64]

__global__ __launch_bounds__(256)
void attn_kernel(const u16* __restrict__ q, const u16* __restrict__ k,
                 const u16* __restrict__ v, u16* __restrict__ o)
{
  __shared__ u16 lK[BC * LDK_];
  __shared__ u16 lV[HD_ * LDV_];
  __shared__ u16 lP[8 * 16 * LDP_];

  const int h     = blockIdx.x;
  const int qb    = blockIdx.y;
  const int tid   = threadIdx.x;
  const int lane  = tid & 31;
  const int wave  = tid >> 5;
  const int lhalf = lane & 15;
  const bool hi   = lane >= 16;
  const int ka    = hi ? 16 : 0;
  const int kbb   = hi ? 16 : 0;

  const int qrow0 = qb * BR + wave * 16;

  // Q fragments (head dim 128 = 4 WMMA k-steps), loaded once from global
  FragU qf[4];
  {
    const u16* qp = q + (size_t)(qrow0 + lhalf) * D_ + h * HD_;
    #pragma unroll
    for (int kk = 0; kk < 4; ++kk) {
      qf[kk].q[0] = *(const uint4*)(qp + kk*32 + ka);
      qf[kk].q[1] = *(const uint4*)(qp + kk*32 + ka + 8);
    }
  }

  v8f acc[8];
  for (int t = 0; t < 8; ++t) acc[t] = vzero();
  float mrow[8], lrow[8];
  for (int i = 0; i < 8; ++i) { mrow[i] = -__builtin_inff(); lrow[i] = 0.f; }

  const float SCALE = 0.08838834764831845f;   // 1/sqrt(128)
  const int jmax = (qb * BR + BR - 1) / BC;

  for (int j = 0; j <= jmax; ++j) {
    const int kb = j * BC;
    __syncthreads();
    // stage K tile via async copy: lK[key][hd]
    {
      int key = tid & 63, ks = (tid >> 6) * 32;
      const u16* src = k + (size_t)(kb + key) * D_ + h * HD_ + ks;
      u16* dst = &lK[key * LDK_ + ks];
      cp_async_b128(dst,      src);
      cp_async_b128(dst + 8,  src + 8);
      cp_async_b128(dst + 16, src + 16);
      cp_async_b128(dst + 24, src + 24);
    }
    // stage V transposed: lV[hd][key]
    {
      int key = tid & 63, hs = (tid >> 6) * 32;
      union { uint4 u[4]; u16 s[32]; } buf;
      const uint4* src = (const uint4*)(v + (size_t)(kb + key) * D_ + h * HD_ + hs);
      buf.u[0] = src[0]; buf.u[1] = src[1]; buf.u[2] = src[2]; buf.u[3] = src[3];
      #pragma unroll
      for (int i = 0; i < 32; ++i) lV[(hs + i) * LDV_ + key] = buf.s[i];
    }
    wait_async0();
    __syncthreads();

    // scores S = Q * K^T. Flattened t(x4) x kk(x4) pipeline with two
    // B-fragment slots: load frag i+1 before WMMA on frag i.
    float sc[4][8];
    {
      FragU fb[2];
      {
        const uint4* p = (const uint4*)&lK[lhalf * LDK_ + kbb];
        fb[0].q[0] = p[0]; fb[0].q[1] = p[1];
      }
      v8f cc = vzero();
      #pragma unroll
      for (int i = 0; i < 16; ++i) {
        const int t = i >> 2, kk = i & 3;
        if (i < 15) {
          const int tn = (i + 1) >> 2, kn = (i + 1) & 3;
          const uint4* p =
              (const uint4*)&lK[(tn*16 + lhalf) * LDK_ + kn*32 + kbb];
          fb[(i + 1) & 1].q[0] = p[0];
          fb[(i + 1) & 1].q[1] = p[1];
        }
        cc = __builtin_amdgcn_wmma_f32_16x16x32_bf16(
            false, qf[kk].v, false, fb[i & 1].v, (short)0, cc, false, false);
        if (kk == 3) {
          const int key = kb + t*16 + lhalf;
          #pragma unroll
          for (int r = 0; r < 8; ++r) {
            int qi = qrow0 + r + (hi ? 8 : 0);
            sc[t][r] = (key <= qi) ? cc[r] * SCALE : -__builtin_inff();
          }
          cc = vzero();
        }
      }
    }
    // online softmax (row lives across 16 lanes of a half-wave)
    float mnew[8], rs[8];
    for (int r = 0; r < 8; ++r) {
      float mx = fmaxf(fmaxf(sc[0][r], sc[1][r]), fmaxf(sc[2][r], sc[3][r]));
      mx = redmax16(mx);
      mnew[r] = fmaxf(mrow[r], mx);
    }
    u16* myP = &lP[wave * 16 * LDP_];
    for (int r = 0; r < 8; ++r) {
      float rsum = 0.f;
      int prow = r + (hi ? 8 : 0);
      #pragma unroll
      for (int t = 0; t < 4; ++t) {
        float pv = __expf(sc[t][r] - mnew[r]);
        rsum += pv;
        myP[prow * LDP_ + t*16 + lhalf] = f2bf(pv);   // C-layout -> LDS
      }
      rs[r] = redsum16(rsum);
    }
    for (int r = 0; r < 8; ++r) {
      float f = __expf(mrow[r] - mnew[r]);
      lrow[r] = lrow[r] * f + rs[r];
      mrow[r] = mnew[r];
      #pragma unroll
      for (int t = 0; t < 8; ++t) acc[t][r] *= f;
    }
    // O += P * V (same-wave DS write->read is in-order). Both P fragments
    // preloaded; V fragments double-buffered over flattened kk(x2) x nt(x8).
    {
      FragU pf[2];
      #pragma unroll
      for (int kk = 0; kk < 2; ++kk) {
        const u16* pp = &myP[lhalf * LDP_ + kk*32 + ka];
        pf[kk].q[0] = *(const uint4*)(pp);
        pf[kk].q[1] = *(const uint4*)(pp + 8);
      }
      FragU fv[2];
      {
        const uint4* p = (const uint4*)&lV[lhalf * LDV_ + kbb];
        fv[0].q[0] = p[0]; fv[0].q[1] = p[1];
      }
      #pragma unroll
      for (int i = 0; i < 16; ++i) {
        const int kk = i >> 3, nt = i & 7;
        if (i < 15) {
          const int kn = (i + 1) >> 3, nn = (i + 1) & 7;
          const uint4* p =
              (const uint4*)&lV[(nn*16 + lhalf) * LDV_ + kn*32 + kbb];
          fv[(i + 1) & 1].q[0] = p[0];
          fv[(i + 1) & 1].q[1] = p[1];
        }
        acc[nt] = __builtin_amdgcn_wmma_f32_16x16x32_bf16(
            false, pf[kk].v, false, fv[i & 1].v, (short)0, acc[nt], false, false);
      }
    }
  }

  // epilogue: O /= l, store bf16 into attn buffer [s][h*128+d]
  for (int r = 0; r < 8; ++r) {
    float inv = 1.0f / lrow[r];
    int row = qrow0 + r + (hi ? 8 : 0);
    u16* op = o + (size_t)row * D_ + h * HD_ + lhalf;
    #pragma unroll
    for (int nt = 0; nt < 8; ++nt)
      op[nt * 16] = f2bf(acc[nt][r] * inv);
  }
}

// ---------------------------------------------------------------------------
extern "C" void kernel_launch(void* const* d_in, const int* in_sizes, int n_in,
                              void* d_out, int out_size, void* d_ws, size_t ws_size,
                              hipStream_t stream)
{
  (void)in_sizes; (void)n_in; (void)out_size; (void)ws_size;
  const float* x  = (const float*)d_in[0];
  const int*  pos = (const int*)d_in[1];
  const float* Wq = (const float*)d_in[2];
  const float* Wk = (const float*)d_in[3];
  const float* Wv = (const float*)d_in[4];
  const float* Wo = (const float*)d_in[5];
  float* out = (float*)d_out;

  u16* qb = (u16*)d_ws;                 // S*D bf16 each (16 MB)
  u16* kb = qb + (size_t)S_ * D_;
  u16* vb = kb + (size_t)S_ * D_;
  u16* ab = vb + (size_t)S_ * D_;

  dim3 gg(S_ / TM, D_ / TN);            // 16 x 32 blocks
  gemm_bt<false, true><<<gg, 256, 0, stream>>>((const void*)x, Wq, (void*)qb, S_, D_, D_);
  gemm_bt<false, true><<<gg, 256, 0, stream>>>((const void*)x, Wk, (void*)kb, S_, D_, D_);
  gemm_bt<false, true><<<gg, 256, 0, stream>>>((const void*)x, Wv, (void*)vb, S_, D_, D_);

  rope_kernel<<<(S_ * NH_ * 64) / 256, 256, 0, stream>>>(qb, kb, pos);

  attn_kernel<<<dim3(NH_, S_ / BR), 256, 0, stream>>>(qb, kb, vb, ab);

  gemm_bt<true, false><<<gg, 256, 0, stream>>>((const void*)ab, Wo, (void*)out, S_, D_, D_);
}
